// NPUMatmulLayer_63943473102887
// MI455X (gfx1250) — 8-bit matrix operand order repaired and checked by reading, compiled, not run on hardware
//
#include <hip/hip_runtime.h>
#include <cstdint>
#include <cstddef>

typedef __attribute__((ext_vector_type(8))) int v8i;

__global__ void npu_init_max(unsigned int* __restrict__ maxbits) {
    if (threadIdx.x == 0 && blockIdx.x == 0) *maxbits = 0u;
}

__global__ __launch_bounds__(256)
void npu_absmax(const float* __restrict__ x, unsigned int* __restrict__ maxbits, int n4) {
    const float4* x4 = (const float4*)x;
    float m = 0.0f;
    for (int i = blockIdx.x * blockDim.x + threadIdx.x; i < n4; i += gridDim.x * blockDim.x) {
        float4 v = x4[i];
        m = fmaxf(m, fmaxf(fmaxf(fabsf(v.x), fabsf(v.y)), fmaxf(fabsf(v.z), fabsf(v.w))));
    }
    #pragma unroll
    for (int off = 16; off > 0; off >>= 1)
        m = fmaxf(m, __shfl_xor(m, off, 32));

    __shared__ float sm[8];
    int wave = threadIdx.x >> 5;
    if ((threadIdx.x & 31) == 0) sm[wave] = m;
    __syncthreads();
    if (threadIdx.x == 0) {
        float bm = sm[0];
        #pragma unroll
        for (int i = 1; i < 8; ++i) bm = fmaxf(bm, sm[i]);
        atomicMax(maxbits, __float_as_uint(bm));
    }
}

__global__ __launch_bounds__(256)
void npu_quant(const float* __restrict__ x, const unsigned int* __restrict__ maxbits,
               int8_t* __restrict__ xq, int n4) {
    float mv = __uint_as_float(*maxbits);
    float s  = (mv > 0.0f) ? (127.0f / mv) : 1.0f;
    const float4* x4 = (const float4*)x;
    char4* q4 = (char4*)xq;
    for (int i = blockIdx.x * blockDim.x + threadIdx.x; i < n4; i += gridDim.x * blockDim.x) {
        float4 v = x4[i];
        char4 q;
        q.x = (signed char)__float2int_rn(v.x * s);
        q.y = (signed char)__float2int_rn(v.y * s);
        q.z = (signed char)__float2int_rn(v.z * s);
        q.w = (signed char)__float2int_rn(v.w * s);
        q4[i] = q;
    }
}

__global__ __launch_bounds__(256)
void npu_gemm_i8_wmma(const int8_t* __restrict__ A,
                      const int8_t* __restrict__ W,
                      float* __restrict__ out,
                      int M, int N, int K) {
    const int wave = threadIdx.x >> 5;
    const int lane = threadIdx.x & 31;
    const int half = lane >> 4;
    const int l16  = lane & 15;

    const int m_base = blockIdx.x * 128 + (wave & 3) * 32;
    const int n_base = blockIdx.y * 64  + (wave >> 2) * 32;

    v8i acc[2][2];
    #pragma unroll
    for (int i = 0; i < 2; ++i)
        #pragma unroll
        for (int j = 0; j < 2; ++j)
            acc[i][j] = (v8i){0, 0, 0, 0, 0, 0, 0, 0};

    for (int kb = 0; kb < K; kb += 64) {
        v8i a[2];
        #pragma unroll
        for (int fm = 0; fm < 2; ++fm) {
            const int8_t* ap = A + (size_t)(m_base + fm * 16 + l16) * K + kb + half * 16;
            int4 d0 = *(const int4*)(ap +  0);
            int4 d1 = *(const int4*)(ap + 32);
            a[fm] = (v8i){d0.x, d0.y, d0.z, d0.w, d1.x, d1.y, d1.z, d1.w};
        }
        v8i b[2];
        #pragma unroll
        for (int fn = 0; fn < 2; ++fn) {
            const int8_t* bp = W + (size_t)(n_base + fn * 16 + l16) * K + kb + half * 16;
            int4 lo = *(const int4*)(bp +  0);
            int4 hi = *(const int4*)(bp + 32);
            b[fn] = (v8i){lo.x, lo.y, lo.z, lo.w, hi.x, hi.y, hi.z, hi.w};
        }
        #pragma unroll
        for (int fm = 0; fm < 2; ++fm)
            #pragma unroll
            for (int fn = 0; fn < 2; ++fn)
                acc[fm][fn] = __builtin_amdgcn_wmma_i32_16x16x64_iu8(
                     true, a[fm],  true, b[fn],
                    acc[fm][fn],  false,  false);
    }

    #pragma unroll
    for (int fm = 0; fm < 2; ++fm) {
        #pragma unroll
        for (int fn = 0; fn < 2; ++fn) {
            const int row0 = m_base + fm * 16 + half * 8;
            const int col  = n_base + fn * 16 + l16;
            float* op = out + (size_t)row0 * N + col;
            #pragma unroll
            for (int v = 0; v < 8; ++v)
                op[(size_t)v * N] = (float)acc[fm][fn][v];
        }
    }
}

extern "C" void kernel_launch(void* const* d_in, const int* in_sizes, int n_in,
                              void* d_out, int out_size, void* d_ws, size_t ws_size,
                              hipStream_t stream) {
    const float*  x = (const float*)d_in[0];
    const int8_t* w = (const int8_t*)d_in[1];
    float* out = (float*)d_out;

    const int K = 1024;
    const int N = 1024;
    const int M = in_sizes[0] / K;

    unsigned int* maxbits = (unsigned int*)d_ws;
    int8_t* xq = (int8_t*)d_ws + 256;

    const int n4 = (M * K) / 4;

    npu_init_max<<<1, 64, 0, stream>>>(maxbits);
    npu_absmax<<<1024, 256, 0, stream>>>(x, maxbits, n4);
    npu_quant<<<2048, 256, 0, stream>>>(x, maxbits, xq, n4);

    dim3 grid(M / 128, N / 64);
    npu_gemm_i8_wmma<<<grid, 256, 0, stream>>>(xq, w, out, M, N, K);
}
